// SLSTM_KAN_84353157694100
// MI455X (gfx1250) — matrix operand order repaired and checked by reading, not compiled, not run on hardware
//
#include <hip/hip_runtime.h>
#include <hip/hip_bf16.h>

// ---------------------------------------------------------------------------
// Problem constants (from reference): B=256, T=512, F=512, D=512, NS=8
// ---------------------------------------------------------------------------
#define BB   256
#define TT   512
#define FF   512
#define DD   512
#define NS   8
#define MROWS (BB * TT)        // 131072
#define N4   (4 * DD)          // 2048

typedef __attribute__((ext_vector_type(16))) __bf16 bf16x16;
typedef __attribute__((ext_vector_type(8)))  __bf16 bf16x8;
typedef __attribute__((ext_vector_type(8)))  float  f32x8;
typedef __attribute__((ext_vector_type(4)))  int    v4i;

// Optional CDNA5 async global->LDS path (guarded: falls back to plain copies)
#if defined(__has_builtin)
#if __has_builtin(__builtin_amdgcn_global_load_async_to_lds_b128) && \
    __has_builtin(__builtin_amdgcn_s_wait_asynccnt)
#define USE_ASYNC_LDS 1
#endif
#endif

// ---------------------------------------------------------------------------
// WMMA helpers: v_wmma_f32_16x16x32_bf16 fragment loads per CDNA5 ISA layouts
// ---------------------------------------------------------------------------
__device__ __forceinline__ f32x8 wmma_bf16(bf16x16 a, bf16x16 b, f32x8 c) {
  return __builtin_amdgcn_wmma_f32_16x16x32_bf16(
      /*neg_a=*/false, a, /*neg_b=*/false, b,
      /*c_mod=*/(short)0, c, /*reuse_a=*/false, /*reuse_b=*/false);
}

// A: 16x32 bf16, row-major source [.., lda] in K. Per ISA:
// lane<16: row=lane,    elems 0..7 -> K=k0..k0+7,  elems 8..15 -> K=k0+16..k0+23
// lane>=16: row=lane-16, elems 0..7 -> K=k0+8..+15, elems 8..15 -> K=k0+24..+31
__device__ __forceinline__ bf16x16 load_a_frag(const __bf16* __restrict__ A,
                                               long lda, long m0, int k0, int lane) {
  int r   = lane & 15;
  int klo = k0 + ((lane & 16) ? 8 : 0);
  const __bf16* p = A + (size_t)(m0 + r) * (size_t)lda + klo;
  bf16x8 lo = *(const bf16x8*)p;
  bf16x8 hi = *(const bf16x8*)(p + 16);
  return __builtin_shufflevector(lo, hi, 0, 1, 2, 3, 4, 5, 6, 7,
                                 8, 9, 10, 11, 12, 13, 14, 15);
}

// B: 32x16 bf16 from pre-transposed weights BT[n, k] (ldb = K). Per ISA:
// lanes 0-15 hold K=k0..k0+7 (elems 0..7), K=k0+16..k0+23 (elems 8..15) of column n,
// lanes 16-31 hold K=k0+8..k0+15, K=k0+24..k0+31.
__device__ __forceinline__ bf16x16 load_b_frag(const __bf16* __restrict__ BT,
                                               long ldb, int n, int k0, int lane) {
  int kk = k0 + ((lane & 16) ? 8 : 0);
  const __bf16* p = BT + (size_t)n * (size_t)ldb + kk;
  bf16x8 lo = *(const bf16x8*)p;
  bf16x8 hi = *(const bf16x8*)(p + 16);
  return __builtin_shufflevector(lo, hi, 0, 1, 2, 3, 4, 5, 6, 7,
                                 8, 9, 10, 11, 12, 13, 14, 15);
}

// Same B fragment but from an LDS-cached slice with padded pitch (520 bf16:
// row stride = 260 dwords == 4 mod 64 banks -> conflict-free b128 reads).
#define UPITCH 520
__device__ __forceinline__ bf16x16 load_b_frag_lds(const __bf16* u, int col,
                                                   int k0, int lane) {
  int kk = k0 + ((lane & 16) ? 8 : 0);
  const __bf16* p = u + (size_t)col * UPITCH + kk;
  bf16x8 lo = *(const bf16x8*)p;
  bf16x8 hi = *(const bf16x8*)(p + 16);
  return __builtin_shufflevector(lo, hi, 0, 1, 2, 3, 4, 5, 6, 7,
                                 8, 9, 10, 11, 12, 13, 14, 15);
}

// ---------------------------------------------------------------------------
// Activation helpers
// ---------------------------------------------------------------------------
__device__ __forceinline__ float sigmoid_f(float x) {
  return 1.0f / (1.0f + __expf(-x));
}
__device__ __forceinline__ float gelu_tanh_f(float x) {
  const float k = 0.7978845608028654f;  // sqrt(2/pi)
  float t = tanhf(k * (x + 0.044715f * x * x * x));
  return 0.5f * x * (1.0f + t);
}
__device__ __forceinline__ float kan_act(int n, float x) {
  switch (n & 3) {
    case 0:  return fmaxf(x, 0.0f);  // relu
    case 1:  return tanhf(x);        // tanh
    case 2:  return sigmoid_f(x);    // sigmoid
    default: return gelu_tanh_f(x);  // gelu (tanh approx, jax default)
  }
}

// ---------------------------------------------------------------------------
// Kernel 0: zero-init barrier array, h ping-pong, c state
// ---------------------------------------------------------------------------
__global__ void init_state_kernel(__bf16* hA, __bf16* hB, float* c,
                                  unsigned* bar) {
  long i = (long)blockIdx.x * blockDim.x + threadIdx.x;
  if (i < (long)BB * DD) {
    hA[i] = (__bf16)0.0f;
    hB[i] = (__bf16)0.0f;
    c[i]  = 0.0f;
  }
  if (i < TT) bar[i] = 0u;
}

// ---------------------------------------------------------------------------
// Kernel 1: transpose fp32 [K,N] weight -> bf16 [N,K]  (Wx and Uh, 2MB each)
// ---------------------------------------------------------------------------
__global__ void pack_weightT_kernel(const float* __restrict__ src,
                                    __bf16* __restrict__ dst, int K, int N) {
  long i = (long)blockIdx.x * blockDim.x + threadIdx.x;  // over N*K
  if (i >= (long)K * N) return;
  int n = (int)(i / K);
  int k = (int)(i % K);
  dst[i] = (__bf16)src[(long)k * N + n];
}

// ---------------------------------------------------------------------------
// Kernel 2: fused KAN: raw -> activations -> agg, output bf16 [MROWS, D]
// One wave (32 lanes) per row; 8 waves per block.
// ---------------------------------------------------------------------------
__global__ __launch_bounds__(256)
void kan_agg_kernel(const float* __restrict__ in,
                    const float* __restrict__ subW,
                    const float* __restrict__ subB,
                    const float* __restrict__ aggW,
                    __bf16* __restrict__ agg) {
  int lane = threadIdx.x & 31;
  int wave = threadIdx.x >> 5;
  long row = (long)blockIdx.x * 8 + wave;  // 0 .. MROWS-1

  const float* x = in + row * FF + lane * 16;
  float xv[16];
#pragma unroll
  for (int j = 0; j < 16; ++j) xv[j] = x[j];

  float p[NS];
#pragma unroll
  for (int n = 0; n < NS; ++n) {
    float s = 0.0f;
    const float* w = subW + n * FF + lane * 16;
#pragma unroll
    for (int j = 0; j < 16; ++j) s = fmaf(xv[j], w[j], s);
    p[n] = s;
  }
  // butterfly reduce across the wave (all lanes end with the full sum)
#pragma unroll
  for (int n = 0; n < NS; ++n) {
    float v = p[n];
#pragma unroll
    for (int off = 16; off > 0; off >>= 1) v += __shfl_xor(v, off, 32);
    p[n] = v + subB[n];
  }
  float a[NS];
#pragma unroll
  for (int n = 0; n < NS; ++n) a[n] = kan_act(n, p[n]);

  // agg row: each lane produces 16 of the 512 outputs
  bf16x16 o;
#pragma unroll
  for (int j = 0; j < 16; ++j) {
    int d = lane * 16 + j;
    float s = 0.0f;
#pragma unroll
    for (int n = 0; n < NS; ++n) s = fmaf(a[n], aggW[n * DD + d], s);
    o[j] = (__bf16)s;
  }
  *(bf16x16*)(agg + row * DD + lane * 16) = o;
}

// ---------------------------------------------------------------------------
// Kernel 3: gx = agg @ Wx + b   (M=131072, N=2048, K=512), bf16 WMMA GEMM.
// Block = 256 threads (8 waves, 2x4 wave grid), WG tile 64x128,
// wave tile 32x32 (2x2 WMMA tiles). Wx^T (2 MB) is L2-resident.
// ---------------------------------------------------------------------------
__global__ __launch_bounds__(256)
void gx_gemm_kernel(const __bf16* __restrict__ agg,
                    const __bf16* __restrict__ WxT,
                    const float* __restrict__ bias,
                    __bf16* __restrict__ gx) {
  int lane = threadIdx.x & 31;
  int wave = threadIdx.x >> 5;
  int wr = (wave >> 2) * 32;         // 0 or 32
  int wc = (wave & 3) * 32;          // 0,32,64,96
  long m0 = (long)blockIdx.y * 64;   // M tile
  int n0 = blockIdx.x * 128;         // N tile

  f32x8 acc[2][2];
#pragma unroll
  for (int i = 0; i < 2; ++i)
#pragma unroll
    for (int j = 0; j < 2; ++j) acc[i][j] = (f32x8)(0.0f);

  for (int k0 = 0; k0 < FF; k0 += 32) {
    bf16x16 a0 = load_a_frag(agg, FF, m0 + wr, k0, lane);
    bf16x16 a1 = load_a_frag(agg, FF, m0 + wr + 16, k0, lane);
    int nA = n0 + wc + (lane & 15);
    bf16x16 b0 = load_b_frag(WxT, FF, nA, k0, lane);
    bf16x16 b1 = load_b_frag(WxT, FF, nA + 16, k0, lane);
    acc[0][0] = wmma_bf16(a0, b0, acc[0][0]);
    acc[0][1] = wmma_bf16(a0, b1, acc[0][1]);
    acc[1][0] = wmma_bf16(a1, b0, acc[1][0]);
    acc[1][1] = wmma_bf16(a1, b1, acc[1][1]);
  }

  // Epilogue: element e of C frag -> (m = e + 8*(lane>=16), n = lane&15)
  int r8 = (lane & 16) ? 8 : 0;
  int nl = lane & 15;
#pragma unroll
  for (int i = 0; i < 2; ++i)
#pragma unroll
    for (int j = 0; j < 2; ++j) {
      long gmBase = m0 + wr + i * 16 + r8;
      int gn = n0 + wc + j * 16 + nl;
      float bv = bias[gn];
#pragma unroll
      for (int e = 0; e < 8; ++e) {
        gx[(gmBase + e) * (long)N4 + gn] = (__bf16)(acc[i][j][e] + bv);
      }
    }
}

// ---------------------------------------------------------------------------
// Kernel 4: persistent sLSTM recurrence. 32 WGs:
//   blockIdx.x>>4 = m-tile (2 tiles of 128 rows over B=256),
//   blockIdx.x&15 = d-chunk (16 chunks of 32 over D=512).
// The WG's 128-column Uh^T slice (4 gates x 32 d-cols x K=512, 128 KB) is
// staged ONCE into LDS (async global->LDS when available) and reused for all
// 512 steps. Per step: g[128x128] = h @ Uh_slice (A from L2, B from LDS),
// gates fused via LDS gbuf, c (fp32, WG-private) and h (bf16 ping-pong)
// updated, then an atomic grid barrier releases step t+1.
// ---------------------------------------------------------------------------
#define NWG_LSTM 32

__global__ __launch_bounds__(256)
void slstm_kernel(const __bf16* __restrict__ gx,
                  const __bf16* __restrict__ UhT,
                  __bf16* __restrict__ hA, __bf16* __restrict__ hB,
                  float* __restrict__ c, unsigned* __restrict__ bar,
                  float* __restrict__ out) {
  int lane = threadIdx.x & 31;
  int wave = threadIdx.x >> 5;
  int wr = (wave >> 1) * 32;   // 0,32,64,96  (4 wave-rows)
  int wc = (wave & 1) * 64;    // 0,64        (2 wave-cols, 64 wide each)
  int m0 = (blockIdx.x >> 4) * 128;
  int d0 = (blockIdx.x & 15) * 32;

  __shared__ __align__(16) __bf16 ubuf[128 * UPITCH];  // 130 KB Uh slice
  __shared__ float gbuf[128 * 128];                    // 64 KB gate buffer

  // ---- stage Uh^T slice into LDS (once; constant over all timesteps) ----
  // tile col j (0..127) -> Uh column n = gate(j>>5)*512 + d0 + (j&31)
  for (int i = threadIdx.x; i < 128 * 64; i += 256) {  // 16B segments
    int j = i >> 6;
    int s = i & 63;
    int n = ((j >> 5) << 9) + d0 + (j & 31);
    const __bf16* src = UhT + (size_t)n * DD + s * 8;
    __bf16* dst = &ubuf[j * UPITCH + s * 8];
#if defined(USE_ASYNC_LDS)
    __builtin_amdgcn_global_load_async_to_lds_b128((v4i*)src, (v4i*)dst, 0, 0);
#else
    *(bf16x8*)dst = *(const bf16x8*)src;
#endif
  }
#if defined(USE_ASYNC_LDS)
  __builtin_amdgcn_s_wait_asynccnt(0);
#endif
  __syncthreads();

  int r8 = (lane & 16) ? 8 : 0;
  int nl = lane & 15;

  for (int t = 0; t < TT; ++t) {
    const __bf16* hc = (t & 1) ? hB : hA;  // read state (h_t)
    __bf16* hn       = (t & 1) ? hA : hB;  // write state (h_{t+1})

    f32x8 acc[2][4];
#pragma unroll
    for (int i = 0; i < 2; ++i)
#pragma unroll
      for (int j = 0; j < 4; ++j) acc[i][j] = (f32x8)(0.0f);

    for (int k0 = 0; k0 < DD; k0 += 32) {
      bf16x16 a0 = load_a_frag(hc, DD, m0 + wr, k0, lane);
      bf16x16 a1 = load_a_frag(hc, DD, m0 + wr + 16, k0, lane);
#pragma unroll
      for (int j = 0; j < 4; ++j) {
        bf16x16 b = load_b_frag_lds(ubuf, wc + j * 16 + nl, k0, lane);
        acc[0][j] = wmma_bf16(a0, b, acc[0][j]);
        acc[1][j] = wmma_bf16(a1, b, acc[1][j]);
      }
    }

    // dump C fragments to LDS: gbuf[m_local * 128 + tile_col]
#pragma unroll
    for (int i = 0; i < 2; ++i)
#pragma unroll
      for (int j = 0; j < 4; ++j) {
        int mBase = wr + i * 16 + r8;
        int cCol = wc + j * 16 + nl;
#pragma unroll
        for (int e = 0; e < 8; ++e)
          gbuf[(mBase + e) * 128 + cCol] = acc[i][j][e];
      }
    __syncthreads();

    // fused gate nonlinearity: 128 rows x 32 d per WG
    for (int idx = threadIdx.x; idx < 128 * 32; idx += 256) {
      int mm = idx >> 5;
      int dd = idx & 31;
      long gxrow = ((long)(m0 + mm) * TT + t) * (long)N4;
      int dG = d0 + dd;
      float gi = gbuf[mm * 128 + dd]      + (float)gx[gxrow + 0 * DD + dG];
      float gf = gbuf[mm * 128 + 32 + dd] + (float)gx[gxrow + 1 * DD + dG];
      float go = gbuf[mm * 128 + 64 + dd] + (float)gx[gxrow + 2 * DD + dG];
      float gc = gbuf[mm * 128 + 96 + dd] + (float)gx[gxrow + 3 * DD + dG];
      int ci = (m0 + mm) * DD + dG;
      float cv = sigmoid_f(gf) * c[ci] + sigmoid_f(gi) * tanhf(gc);
      float hv = sigmoid_f(go) * tanhf(cv);
      c[ci] = cv;
      hn[ci] = (__bf16)hv;
      if (t == TT - 1) out[ci] = hv;
    }

    if (t == TT - 1) break;

    // ---- grid barrier across the 32 persistent WGs ----
    __threadfence();        // release this thread's h/c stores (agent scope)
    __syncthreads();        // all threads in WG have released
    if (threadIdx.x == 0) {
      __hip_atomic_fetch_add(bar + t, 1u, __ATOMIC_ACQ_REL,
                             __HIP_MEMORY_SCOPE_AGENT);
      while (__hip_atomic_load(bar + t, __ATOMIC_ACQUIRE,
                               __HIP_MEMORY_SCOPE_AGENT) < NWG_LSTM) {
        __builtin_amdgcn_s_sleep(2);
      }
    }
    __syncthreads();
    __threadfence();        // acquire: invalidate stale cached h
  }
}

// ---------------------------------------------------------------------------
// Launch
// ---------------------------------------------------------------------------
extern "C" void kernel_launch(void* const* d_in, const int* in_sizes, int n_in,
                              void* d_out, int out_size, void* d_ws, size_t ws_size,
                              hipStream_t stream) {
  const float* in_x  = (const float*)d_in[0];  // [B,T,F]
  const float* subW  = (const float*)d_in[1];  // [8,512]
  const float* subB  = (const float*)d_in[2];  // [8]
  const float* aggW  = (const float*)d_in[3];  // [8,512]
  const float* Wx    = (const float*)d_in[4];  // [512,2048]
  const float* Uh    = (const float*)d_in[5];  // [512,2048]
  const float* bias  = (const float*)d_in[6];  // [2048]
  float* out = (float*)d_out;                  // [256,512]

  // scratch carve-out
  char* p = (char*)d_ws;
  auto carve = [&](size_t bytes) {
    void* r = (void*)p;
    p += (bytes + 255) & ~(size_t)255;
    return r;
  };
  unsigned* bar = (unsigned*)carve(TT * sizeof(unsigned));
  __bf16* hA    = (__bf16*)carve((size_t)BB * DD * 2);
  __bf16* hB    = (__bf16*)carve((size_t)BB * DD * 2);
  float*  cst   = (float*) carve((size_t)BB * DD * 4);
  __bf16* WxT   = (__bf16*)carve((size_t)N4 * DD * 2);     // [2048,512]
  __bf16* UhT   = (__bf16*)carve((size_t)N4 * DD * 2);     // [2048,512]
  __bf16* agg   = (__bf16*)carve((size_t)MROWS * DD * 2);  // 128 MB
  __bf16* gx    = (__bf16*)carve((size_t)MROWS * N4 * 2);  // 512 MB

  // 0) init barrier + states (h0 = c0 = 0)
  {
    long n = (long)BB * DD;
    int blocks = (int)((n + 255) / 256);
    init_state_kernel<<<blocks, 256, 0, stream>>>(hA, hB, cst, bar);
  }
  // 1) pack weights (fp32 [K,N] -> bf16 [N,K])
  {
    long n = (long)DD * N4;
    int blocks = (int)((n + 255) / 256);
    pack_weightT_kernel<<<blocks, 256, 0, stream>>>(Wx, WxT, DD, N4);
    pack_weightT_kernel<<<blocks, 256, 0, stream>>>(Uh, UhT, DD, N4);
  }
  // 2) fused KAN -> agg (bf16)
  kan_agg_kernel<<<MROWS / 8, 256, 0, stream>>>(in_x, subW, subB, aggW, agg);
  // 3) gx = agg @ Wx + b  (WMMA bf16)
  {
    dim3 grid(N4 / 128, MROWS / 64);  // (16, 2048)
    gx_gemm_kernel<<<grid, 256, 0, stream>>>(agg, WxT, bias, gx);
  }
  // 4) persistent sLSTM recurrence (WMMA bf16, LDS-resident Uh, grid barrier)
  slstm_kernel<<<NWG_LSTM, 256, 0, stream>>>(gx, UhT, hA, hB, cst, bar, out);
}
